// MarketStateSpace_4045859193507
// MI455X (gfx1250) — compile-verified, matrix operand order checked by reading, not run on hardware
//
#include <hip/hip_runtime.h>
#include <cstdint>
#include <cstddef>

// ---------------------------------------------------------------------------
// Types for CDNA5 WMMA (wave32, 16x16x32 bf16 -> f32)
// ---------------------------------------------------------------------------
typedef __bf16 bf16_t;
typedef __attribute__((ext_vector_type(8)))  bf16_t v8bf;
typedef __attribute__((ext_vector_type(16))) bf16_t v16bf;
typedef __attribute__((ext_vector_type(8)))  float  v8f;
typedef __attribute__((ext_vector_type(2)))  float  f32x2;

// ---------------------------------------------------------------------------
// Workspace layout (bytes).  Needs ~780KB of d_ws.
// ---------------------------------------------------------------------------
static constexpr size_t OFF_FINV  = 0;        // f32 [64][64]
static constexpr size_t OFF_MET   = 16384;    // f32 [64][64]
static constexpr size_t OFF_MEFF  = 32768;    // f32 [64][64]
static constexpr size_t OFF_T1    = 49152;    // f32 [64][64] temp
static constexpr size_t OFF_T2    = 65536;    // f32 [64][64] temp
static constexpr size_t OFF_WFIN  = 81920;    // f32 [64][64]  W_final[k][o]
static constexpr size_t OFF_BFIN  = 98304;    // f32 [64]
static constexpr size_t OFF_CFAC  = 98560;    // f32 [8][8] cos(ph_h-ph_g)/sqrt(8)
static constexpr size_t OFF_KQ    = 102400;   // bf16 [64 n][64 k]  (Bt layouts)
static constexpr size_t OFF_KK    = OFF_KQ   + 8192;
static constexpr size_t OFF_KV    = OFF_KK   + 8192;
static constexpr size_t OFF_KO    = OFF_KV   + 8192;
static constexpr size_t OFF_KU    = OFF_KO   + 8192;
static constexpr size_t OFF_KS    = OFF_KU   + 8192;
static constexpr size_t OFF_KFINV = OFF_KS   + 8192;
static constexpr size_t OFF_KMET  = OFF_KFINV+ 8192;
static constexpr size_t OFF_KWFIN = OFF_KMET + 8192;
static constexpr size_t OFF_KTOPO = 176128;   // bf16 [64 o][768 k]
static constexpr size_t OFF_KCP   = 274432;   // bf16 [64 j][64 n][64 i]  C' blocks
static constexpr size_t WS_NEEDED = OFF_KCP + 64*64*64*2;   // 798720 bytes

static constexpr int Bsz = 65536;

// ---------------------------------------------------------------------------
// WMMA helpers.
// A-matrix layout (16-bit, 16Mx32K): lane L holds row M=L%16, two contiguous
// 8-half segments at K = c..c+7 and 16+c..16+c+7 with c = (L<16 ? 0 : 8).
// B-matrix (32Kx16N) mirrors this with N = L%16.  So both operands load from
// a row-major [row][ldk] half array with two 16B ds/global loads.
// ---------------------------------------------------------------------------
__device__ __forceinline__ v16bf ld_op16(const bf16_t* __restrict__ base, int row,
                                         int ldk, int kstep, int lane) {
  const int c = (lane & 16) ? 8 : 0;
  const bf16_t* p = base + row * ldk + kstep * 32 + c;
  v8bf lo = *(const v8bf*)p;
  v8bf hi = *(const v8bf*)(p + 16);
  return __builtin_shufflevector(lo, hi, 0,1,2,3,4,5,6,7,8,9,10,11,12,13,14,15);
}

__device__ __forceinline__ v8f wmma_bf16(v16bf a, v16bf b, v8f c) {
  return __builtin_amdgcn_wmma_f32_16x16x32_bf16(false, a, false, b, (short)0, c,
                                                 false, false);
}

// C/D layout: lane L, element e -> row M = e + (L<16?0:8), col N = L%16 (+16*nt)
__device__ __forceinline__ void store_tile_bf16(bf16_t* __restrict__ dst, int ld,
                                                const v8f acc[4], int lane) {
  const int n0 = lane & 15;
  const int hi = (lane & 16) ? 8 : 0;
#pragma unroll
  for (int nt = 0; nt < 4; ++nt) {
    const int n = n0 + nt * 16;
#pragma unroll
    for (int e = 0; e < 8; ++e) dst[(e + hi) * ld + n] = (bf16_t)acc[nt][e];
  }
}

__device__ __forceinline__ void init_bias(v8f acc[4], const float* __restrict__ bias,
                                          int lane) {
  const int n0 = lane & 15;
#pragma unroll
  for (int nt = 0; nt < 4; ++nt) {
    const float bv = bias[n0 + nt * 16];
#pragma unroll
    for (int e = 0; e < 8; ++e) acc[nt][e] = bv;
  }
}

// out[16b x 64n] += A(LDS [16][ldk] bf16, first 64 cols) x Bt(global [64n][64k])
__device__ __forceinline__ void gemm64(const bf16_t* __restrict__ As, int ldk,
                                       const bf16_t* __restrict__ Bg, v8f acc[4],
                                       int lane) {
  const int bl = lane & 15;
#pragma unroll
  for (int ks = 0; ks < 2; ++ks) {
    const v16bf a = ld_op16(As, ldk == 96 ? bl : bl, ldk, ks, lane);
#pragma unroll
    for (int nt = 0; nt < 4; ++nt) {
      const v16bf b = ld_op16(Bg, bl + nt * 16, 64, ks, lane);
      acc[nt] = wmma_bf16(a, b, acc[nt]);
    }
  }
}

// ---------------------------------------------------------------------------
// Kernel 1: one workgroup.  Finv, metric, m_eff, folded W_final/b_final,
// cos-phase table, and all bf16 B-matrix conversions.
// ---------------------------------------------------------------------------
__global__ __launch_bounds__(256) void mss_precompute(
    const float* __restrict__ tk,  const float* __restrict__ qw,
    const float* __restrict__ kw,  const float* __restrict__ vw,
    const float* __restrict__ ow,  const float* __restrict__ phase,
    const float* __restrict__ uw,  const float* __restrict__ sw,
    const float* __restrict__ fm,  const float* __restrict__ mm,
    const float* __restrict__ pw,  const float* __restrict__ pb,
    const float* __restrict__ oe,  const float* __restrict__ mor,
    const float* __restrict__ fwv, const float* __restrict__ outw,
    const float* __restrict__ outb, char* __restrict__ ws) {
  float* FINV = (float*)(ws + OFF_FINV);
  float* MET  = (float*)(ws + OFF_MET);
  float* MEFF = (float*)(ws + OFF_MEFF);
  float* T1   = (float*)(ws + OFF_T1);
  float* T2   = (float*)(ws + OFF_T2);
  float* WFIN = (float*)(ws + OFF_WFIN);
  float* BFIN = (float*)(ws + OFF_BFIN);
  float* CFAC = (float*)(ws + OFF_CFAC);
  bf16_t* KQ    = (bf16_t*)(ws + OFF_KQ);
  bf16_t* KK    = (bf16_t*)(ws + OFF_KK);
  bf16_t* KV    = (bf16_t*)(ws + OFF_KV);
  bf16_t* KO    = (bf16_t*)(ws + OFF_KO);
  bf16_t* KU    = (bf16_t*)(ws + OFF_KU);
  bf16_t* KS    = (bf16_t*)(ws + OFF_KS);
  bf16_t* KFI   = (bf16_t*)(ws + OFF_KFINV);
  bf16_t* KME   = (bf16_t*)(ws + OFF_KMET);
  bf16_t* KWF   = (bf16_t*)(ws + OFF_KWFIN);
  bf16_t* KTOPO = (bf16_t*)(ws + OFF_KTOPO);

  const int tid = threadIdx.x;
  __shared__ float aug[64][128];
  __shared__ float fcol[64];
  __shared__ float fwp[64];
  __shared__ float v1[64];
  __shared__ float v2[64];

  // fisher = fm @ fm^T ; augment with identity
  for (int idx = tid; idx < 4096; idx += 256) {
    const int r = idx >> 6, c = idx & 63;
    float s = 0.f;
    for (int k = 0; k < 64; ++k) s += fm[r * 64 + k] * fm[c * 64 + k];
    aug[r][c] = s;
    aug[r][64 + c] = (r == c) ? 1.f : 0.f;
  }
  __syncthreads();
  // Gauss-Jordan (SPD: no pivoting needed)
  for (int p = 0; p < 64; ++p) {
    const float pv = aug[p][p];
    __syncthreads();
    if (tid < 128) aug[p][tid] = aug[p][tid] * (1.f / pv);
    __syncthreads();
    if (tid < 64) fcol[tid] = aug[tid][p];
    __syncthreads();
    for (int idx = tid; idx < 8192; idx += 256) {
      const int r = idx >> 7, c = idx & 127;
      if (r != p) aug[r][c] -= fcol[r] * aug[p][c];
    }
    __syncthreads();
  }
  for (int idx = tid; idx < 4096; idx += 256) {
    const int r = idx >> 6, c = idx & 63;
    FINV[r * 64 + c] = aug[r][64 + c];
  }
  // metric = mm @ mm^T
  for (int idx = tid; idx < 4096; idx += 256) {
    const int r = idx >> 6, c = idx & 63;
    float s = 0.f;
    for (int k = 0; k < 64; ++k) s += mm[r * 64 + k] * mm[c * 64 + k];
    MET[r * 64 + c] = s;
  }
  // softmax(functor_w), redundant per-thread reduction
  if (tid < 64) {
    float m = -1e30f;
    for (int i = 0; i < 64; ++i) m = fmaxf(m, fwv[i]);
    float s = 0.f;
    for (int i = 0; i < 64; ++i) s += __expf(fwv[i] - m);
    fwp[tid] = __expf(fwv[tid] - m) / s;
  }
  __syncthreads();
  // m_eff = sum_m fw[m] * morphisms[m]
  for (int idx = tid; idx < 4096; idx += 256) {
    const int i = idx >> 6, j = idx & 63;
    float s = 0.f;
    for (int m = 0; m < 64; ++m) s += fwp[m] * mor[((size_t)m * 64 + i) * 64 + j];
    MEFF[i * 64 + j] = s;
  }
  __syncthreads();
  // T1 = proj_w^T @ obj_emb
  for (int idx = tid; idx < 4096; idx += 256) {
    const int k = idx >> 6, c = idx & 63;
    float s = 0.f;
    for (int n = 0; n < 64; ++n) s += pw[n * 64 + k] * oe[n * 64 + c];
    T1[k * 64 + c] = s;
  }
  __syncthreads();
  // T2 = T1 @ m_eff
  for (int idx = tid; idx < 4096; idx += 256) {
    const int k = idx >> 6, c = idx & 63;
    float s = 0.f;
    for (int t = 0; t < 64; ++t) s += T1[k * 64 + t] * MEFF[t * 64 + c];
    T2[k * 64 + c] = s;
  }
  __syncthreads();
  // W_final = T2 @ out_w^T
  for (int idx = tid; idx < 4096; idx += 256) {
    const int k = idx >> 6, o = idx & 63;
    float s = 0.f;
    for (int c = 0; c < 64; ++c) s += T2[k * 64 + c] * outw[o * 64 + c];
    WFIN[k * 64 + o] = s;
  }
  // b_final chain
  if (tid < 64) {
    float s = 0.f;
    for (int n = 0; n < 64; ++n) s += pb[n] * oe[n * 64 + tid];
    v1[tid] = s;
  }
  __syncthreads();
  if (tid < 64) {
    float s = 0.f;
    for (int t = 0; t < 64; ++t) s += v1[t] * MEFF[t * 64 + tid];
    v2[tid] = s;
  }
  __syncthreads();
  if (tid < 64) {
    float s = 0.f;
    for (int c = 0; c < 64; ++c) s += v2[c] * outw[tid * 64 + c];
    BFIN[tid] = s + outb[tid];
  }
  // phase table: cos(ph_h - ph_g)/sqrt(8)
  if (tid < 64) {
    const int h = tid >> 3, g = tid & 7;
    CFAC[tid] = __cosf(phase[h] - phase[g]) * 0.35355339059327373f;
  }
  __syncthreads();
  // bf16 B-matrix conversions (Bt[n][k] layouts)
  for (int idx = tid; idx < 4096; idx += 256) {
    const int n = idx >> 6, k = idx & 63;
    KQ[idx]  = (bf16_t)qw[idx];
    KK[idx]  = (bf16_t)kw[idx];
    KV[idx]  = (bf16_t)vw[idx];
    KO[idx]  = (bf16_t)ow[idx];
    KU[idx]  = (bf16_t)uw[n * 128 + k];   // h0 == 0 -> only first 64 cols matter
    KS[idx]  = (bf16_t)sw[n * 128 + k];
    KFI[idx] = (bf16_t)FINV[k * 64 + n];
    KME[idx] = (bf16_t)MET[k * 64 + n];
    KWF[idx] = (bf16_t)WFIN[k * 64 + n];
  }
  for (int idx = tid; idx < 49152; idx += 256) KTOPO[idx] = (bf16_t)tk[idx];
}

// ---------------------------------------------------------------------------
// Kernel 2: C'[j][n][i] = bf16( sum_k connection[i][j][k] * W_final[k][n] )
// ---------------------------------------------------------------------------
__global__ __launch_bounds__(256) void mss_build_cprime(
    const float* __restrict__ conn, const float* __restrict__ WFIN,
    bf16_t* __restrict__ KCP) {
  __shared__ float Wsh[64][64];
  const int j = blockIdx.x;
  for (int idx = threadIdx.x; idx < 4096; idx += 256)
    Wsh[idx >> 6][idx & 63] = WFIN[idx];
  __syncthreads();
  for (int idx = threadIdx.x; idx < 4096; idx += 256) {
    const int n = idx >> 6, i = idx & 63;
    const float* cr = conn + ((size_t)i * 64 + j) * 64;
    float s = 0.f;
    for (int k = 0; k < 64; ++k) s += cr[k] * Wsh[k][n];
    KCP[((size_t)j * 64 + n) * 64 + i] = (bf16_t)s;
  }
}

// ---------------------------------------------------------------------------
// Main kernel: 1024 blocks x 64 threads (2 wave32).  Each wave carries TWO
// 16-row tiles; the per-tile pipeline runs strictly sequentially (unroll 1)
// so registers are reused, and waves_per_eu(4) caps allocation at 256 VGPRs
// (no VGPR-MSB mode, >=4 waves/SIMD of latency hiding for the HBM phase).
// The C'-loop shares each B-operand pair across both tiles (load:wmma 1:1).
// ---------------------------------------------------------------------------
__global__ __launch_bounds__(64)
__attribute__((amdgpu_waves_per_eu(4)))
void mss_main(
    const float* __restrict__ pers,
    const float* __restrict__ q_b, const float* __restrict__ k_b,
    const float* __restrict__ v_b, const float* __restrict__ o_b,
    const float* __restrict__ u_b, const float* __restrict__ s_b,
    const char* __restrict__ ws, float* __restrict__ out) {
  const bf16_t* KTOPO = (const bf16_t*)(ws + OFF_KTOPO);
  const bf16_t* KQ    = (const bf16_t*)(ws + OFF_KQ);
  const bf16_t* KK    = (const bf16_t*)(ws + OFF_KK);
  const bf16_t* KV    = (const bf16_t*)(ws + OFF_KV);
  const bf16_t* KO    = (const bf16_t*)(ws + OFF_KO);
  const bf16_t* KU    = (const bf16_t*)(ws + OFF_KU);
  const bf16_t* KS    = (const bf16_t*)(ws + OFF_KS);
  const bf16_t* KFI   = (const bf16_t*)(ws + OFF_KFINV);
  const bf16_t* KME   = (const bf16_t*)(ws + OFF_KMET);
  const bf16_t* KWF   = (const bf16_t*)(ws + OFF_KWFIN);
  const bf16_t* KCP   = (const bf16_t*)(ws + OFF_KCP);
  const float*  BFIN  = (const float*)(ws + OFF_BFIN);
  const float*  CFAC  = (const float*)(ws + OFF_CFAC);

  // [wave][tile] private regions
  __shared__ __align__(16) bf16_t Abuf[2][2][16][96];    // A tiles (bf16)
  __shared__ __align__(16) bf16_t QKV[2][2][3][16][64];  // q,k,v (bf16)
  __shared__ __align__(16) float  Xw[2][2][16][64];      // xm (fp32 scaling)
  __shared__ float cfs[64];

  const int tid  = threadIdx.x;
  const int lane = tid & 31;
  const int wv   = tid >> 5;               // 0..1
  const int bl   = lane & 15;
  const int hi8  = (lane & 16) ? 8 : 0;
  const int jh   = lane >> 4;
  const int btw  = blockIdx.x * 64 + wv * 32;   // wave's 32-row base

  cfs[tid] = CFAC[tid];                    // 64 threads == 64 entries

  const f32x2* pp = (const f32x2*)pers;    // (d,t)-pairs: index ((j*B+b)*3+d)

  // ======== per-tile pipeline: topo -> attention -> GRU -> xm ========
#pragma unroll 1
  for (int tt = 0; tt < 2; ++tt) {
    bf16_t* Aw = &Abuf[wv][tt][0][0];
    const int bt = btw + tt * 16;
    const int gb = bt + bl;

    // ---- Stage 1: topo[b,o] = sum_{j,d} (pers summed over t) * tk[o,j,d]
    v8f acc[4];
#pragma unroll
    for (int nt = 0; nt < 4; ++nt)
#pragma unroll
      for (int e = 0; e < 8; ++e) acc[nt][e] = 0.f;

#pragma unroll 1
    for (int jc = 0; jc < 8; ++jc) {
#pragma unroll 4
      for (int jj = 0; jj < 16; ++jj) {
        const int j = jc * 32 + jh * 16 + jj;
        const size_t base = ((size_t)j * Bsz + gb) * 3;
        const f32x2 p0 = pp[base + 0];
        const f32x2 p1 = pp[base + 1];
        const f32x2 p2 = pp[base + 2];
        const int kk = (jh * 16 + jj) * 3;
        Abuf[wv][tt][bl][kk + 0] = (bf16_t)(p0.x + p0.y);
        Abuf[wv][tt][bl][kk + 1] = (bf16_t)(p1.x + p1.y);
        Abuf[wv][tt][bl][kk + 2] = (bf16_t)(p2.x + p2.y);
      }
      __syncthreads();
#pragma unroll
      for (int ks = 0; ks < 3; ++ks) {
        const v16bf a = ld_op16(Aw, bl, 96, ks, lane);
#pragma unroll
        for (int nt = 0; nt < 4; ++nt) {
          const v16bf b = ld_op16(KTOPO, bl + nt * 16, 768, jc * 3 + ks, lane);
          acc[nt] = wmma_bf16(a, b, acc[nt]);
        }
      }
      __syncthreads();
    }
    store_tile_bf16(Aw, 96, acc, lane);    // topo -> A tile
    __syncthreads();

    // ---- q, k, v
    {
      const bf16_t* Bm[3] = {KQ, KK, KV};
      const float*  bs[3] = {q_b, k_b, v_b};
#pragma unroll 1
      for (int m = 0; m < 3; ++m) {
        v8f t4[4];
        init_bias(t4, bs[m], lane);
        gemm64(Aw, 96, Bm[m], t4, lane);
        store_tile_bf16(&QKV[wv][tt][m][0][0], 64, t4, lane);
      }
    }
    __syncthreads();

    // ---- attention (only the real softmax survives): 4 (b,h) rows / lane
#pragma unroll 1
    for (int r = 0; r < 4; ++r) {
      const int row = lane * 4 + r;        // 0..127
      const int b2 = row >> 3, h = row & 7;
      float q8[8];
      {
        const v8bf qv = *(const v8bf*)&QKV[wv][tt][0][b2][h * 8];
#pragma unroll
        for (int d = 0; d < 8; ++d) q8[d] = (float)qv[d];
      }
      float S[8];
      float mx = -1e30f;
#pragma unroll
      for (int g = 0; g < 8; ++g) {
        const v8bf kv = *(const v8bf*)&QKV[wv][tt][1][b2][g * 8];
        float dot = 0.f;
#pragma unroll
        for (int d = 0; d < 8; ++d) dot += q8[d] * (float)kv[d];
        S[g] = dot * cfs[h * 8 + g];
        mx = fmaxf(mx, S[g]);
      }
      float se = 0.f;
#pragma unroll
      for (int g = 0; g < 8; ++g) { S[g] = __expf(S[g] - mx); se += S[g]; }
      const float inv = __builtin_amdgcn_rcpf(se);
      float av[8];
#pragma unroll
      for (int d = 0; d < 8; ++d) av[d] = 0.f;
#pragma unroll
      for (int g = 0; g < 8; ++g) {
        const v8bf vv = *(const v8bf*)&QKV[wv][tt][2][b2][g * 8];
#pragma unroll
        for (int d = 0; d < 8; ++d) av[d] += S[g] * (float)vv[d];
      }
#pragma unroll
      for (int d = 0; d < 8; ++d)
        Abuf[wv][tt][b2][h * 8 + d] = (bf16_t)(av[d] * inv);
    }
    __syncthreads();

    // ---- quantum = av @ o_w^T + o_b
    v8f qa[4];
    init_bias(qa, o_b, lane);
    gemm64(Aw, 96, KO, qa, lane);
    store_tile_bf16(Aw, 96, qa, lane);
    __syncthreads();

    // ---- GRU step, h0=0:  new_h = sigmoid(u) * (tanh(cand) @ Finv)
    v8f ua[4];
    init_bias(ua, u_b, lane);
    gemm64(Aw, 96, KU, ua, lane);
#pragma unroll
    for (int nt = 0; nt < 4; ++nt)
#pragma unroll
      for (int e = 0; e < 8; ++e)
        ua[nt][e] = __builtin_amdgcn_rcpf(1.f + __expf(-ua[nt][e]));

    v8f ca[4];
    init_bias(ca, s_b, lane);
    gemm64(Aw, 96, KS, ca, lane);
#pragma unroll
    for (int nt = 0; nt < 4; ++nt)
#pragma unroll
      for (int e = 0; e < 8; ++e) {
        // tanh(x) = (e^{2x}-1)/(e^{2x}+1), fast-rcp form
        const float t = __expf(2.f * ca[nt][e]);
        ca[nt][e] = (t - 1.f) * __builtin_amdgcn_rcpf(t + 1.f);
      }
    __syncthreads();
    store_tile_bf16(Aw, 96, ca, lane);
    __syncthreads();

    v8f na[4];
#pragma unroll
    for (int nt = 0; nt < 4; ++nt)
#pragma unroll
      for (int e = 0; e < 8; ++e) na[nt][e] = 0.f;
    gemm64(Aw, 96, KFI, na, lane);
#pragma unroll
    for (int nt = 0; nt < 4; ++nt)
#pragma unroll
      for (int e = 0; e < 8; ++e) na[nt][e] *= ua[nt][e];
    __syncthreads();
    store_tile_bf16(Aw, 96, na, lane);     // new_h
    __syncthreads();

    // ---- xm = new_h @ metric
    v8f xa[4];
#pragma unroll
    for (int nt = 0; nt < 4; ++nt)
#pragma unroll
      for (int e = 0; e < 8; ++e) xa[nt][e] = 0.f;
    gemm64(Aw, 96, KME, xa, lane);
    __syncthreads();
    store_tile_bf16(Aw, 96, xa, lane);     // xm (bf16 A operand)
#pragma unroll
    for (int nt = 0; nt < 4; ++nt) {
      const int n = bl + nt * 16;
#pragma unroll
      for (int e = 0; e < 8; ++e) Xw[wv][tt][e + hi8][n] = xa[nt][e];  // fp32
    }
    __syncthreads();
  }

  // ======== out = xm @ W_final + b_final + sum_j xm[:,j]*(xm @ C'_j) ========
  v8f oa[2][4];
#pragma unroll
  for (int tt = 0; tt < 2; ++tt) {
    init_bias(oa[tt], BFIN, lane);
    gemm64(&Abuf[wv][tt][0][0], 96, KWF, oa[tt], lane);
  }

  v16bf a0[2], a1[2];                      // xm operands, j-invariant
#pragma unroll
  for (int tt = 0; tt < 2; ++tt) {
    a0[tt] = ld_op16(&Abuf[wv][tt][0][0], bl, 96, 0, lane);
    a1[tt] = ld_op16(&Abuf[wv][tt][0][0], bl, 96, 1, lane);
  }
  v8f vz;
#pragma unroll
  for (int e = 0; e < 8; ++e) vz[e] = 0.f;

#pragma unroll 1
  for (int j = 0; j < 64; ++j) {
    const bf16_t* Bj = KCP + (size_t)j * 4096;
    // pull next j's C' block toward the WGP (speculative, L2-hot)
    __builtin_prefetch(Bj + (j < 63 ? 4096 : 0), 0, 3);
    float xj0[8], xj1[8];
#pragma unroll
    for (int e = 0; e < 8; ++e) {
      xj0[e] = Xw[wv][0][e + hi8][j];
      xj1[e] = Xw[wv][1][e + hi8][j];
    }
#pragma unroll
    for (int nt = 0; nt < 4; ++nt) {
      const v16bf b0 = ld_op16(Bj, bl + nt * 16, 64, 0, lane);
      const v16bf b1 = ld_op16(Bj, bl + nt * 16, 64, 1, lane);
      v8f t = wmma_bf16(a0[0], b0, vz);
      t = wmma_bf16(a1[0], b1, t);
#pragma unroll
      for (int e = 0; e < 8; ++e) oa[0][nt][e] = fmaf(xj0[e], t[e], oa[0][nt][e]);
      v8f u = wmma_bf16(a0[1], b0, vz);
      u = wmma_bf16(a1[1], b1, u);
#pragma unroll
      for (int e = 0; e < 8; ++e) oa[1][nt][e] = fmaf(xj1[e], u[e], oa[1][nt][e]);
    }
  }

  // ---- store [B,64] fp32
#pragma unroll
  for (int tt = 0; tt < 2; ++tt) {
    const int bt = btw + tt * 16;
#pragma unroll
    for (int nt = 0; nt < 4; ++nt) {
      const int n = bl + nt * 16;
#pragma unroll
      for (int e = 0; e < 8; ++e)
        out[(size_t)(bt + e + hi8) * 64 + n] = oa[tt][nt][e];
    }
  }
}

// ---------------------------------------------------------------------------
extern "C" void kernel_launch(void* const* d_in, const int* in_sizes, int n_in,
                              void* d_out, int out_size, void* d_ws, size_t ws_size,
                              hipStream_t stream) {
  (void)in_sizes; (void)n_in; (void)out_size; (void)ws_size;  // needs >= WS_NEEDED (~780KB)
  char* ws = (char*)d_ws;

  mss_precompute<<<1, 256, 0, stream>>>(
      (const float*)d_in[2],  // topo_kernel
      (const float*)d_in[3],  // q_w
      (const float*)d_in[5],  // k_w
      (const float*)d_in[7],  // v_w
      (const float*)d_in[9],  // o_w
      (const float*)d_in[11], // phase
      (const float*)d_in[12], // update_w
      (const float*)d_in[16], // state_w
      (const float*)d_in[18], // fisher_m
      (const float*)d_in[19], // metric_m
      (const float*)d_in[21], // proj_w
      (const float*)d_in[22], // proj_b
      (const float*)d_in[23], // obj_emb
      (const float*)d_in[24], // morphisms
      (const float*)d_in[25], // functor_w
      (const float*)d_in[26], // out_w
      (const float*)d_in[27], // out_b
      ws);

  mss_build_cprime<<<64, 256, 0, stream>>>(
      (const float*)d_in[20],                // connection
      (const float*)(ws + OFF_WFIN),
      (bf16_t*)(ws + OFF_KCP));

  mss_main<<<Bsz / 64, 64, 0, stream>>>(
      (const float*)d_in[1],  // persistence
      (const float*)d_in[4],  // q_b
      (const float*)d_in[6],  // k_b
      (const float*)d_in[8],  // v_b
      (const float*)d_in[10], // o_b
      (const float*)d_in[13], // update_b
      (const float*)d_in[17], // state_b
      ws, (float*)d_out);
}
